// Transform_Net_89146341196037
// MI455X (gfx1250) — matrix operand order repaired and checked by reading, not compiled, not run on hardware
//
#include <hip/hip_runtime.h>
#include <hip/hip_bf16.h>

typedef __attribute__((ext_vector_type(16))) _Float16 v16h;
typedef __attribute__((ext_vector_type(8)))  _Float16 v8h;
typedef __attribute__((ext_vector_type(8)))  float    v8f;

#define EPS 1e-5f
#define NB 32
#define NPTS 2048
#define KNN 20
#define PTS 8   // points per workgroup in edgeconv

// ---------------------------------------------------------------------------
// Kernel 1: kNN. One block = (batch, 256-point slab). Batch's x (3x2048) and
// squared norms live in LDS; each thread streams 2048 distances and keeps a
// register-resident sorted top-20 (swap-chain insertion, fully unrolled).
// pd is never materialized (saves 537 MB of HBM traffic).
// ---------------------------------------------------------------------------
__global__ __launch_bounds__(256) void knn_kernel(const float* __restrict__ x,
                                                  int* __restrict__ idx_out) {
  __shared__ float xl[3][NPTS];
  __shared__ float xx[NPTS];
  const int tid = threadIdx.x;
  const int b = blockIdx.x >> 3;
  const int i0 = (blockIdx.x & 7) * 256;
  const float* xb = x + (size_t)b * 3 * NPTS;
  for (int t = tid; t < 3 * NPTS; t += 256) xl[t >> 11][t & (NPTS - 1)] = xb[t];
  __syncthreads();
  for (int t = tid; t < NPTS; t += 256) {
    float a0 = xl[0][t], a1 = xl[1][t], a2 = xl[2][t];
    xx[t] = a0 * a0 + a1 * a1 + a2 * a2;
  }
  __syncthreads();
  const int i = i0 + tid;
  const float x0 = xl[0][i], x1 = xl[1][i], x2 = xl[2][i];
  const float xxi = xx[i];
  float bd[KNN]; int bi[KNN];
#pragma unroll
  for (int q = 0; q < KNN; ++q) { bd[q] = -3.4e38f; bi[q] = 0; }
  for (int j = 0; j < NPTS; ++j) {
    float d = 2.0f * (x0 * xl[0][j] + x1 * xl[1][j] + x2 * xl[2][j]) - xxi - xx[j];
    if (d > bd[KNN - 1]) {
      float cd = d; int ci = j;
#pragma unroll
      for (int q = 0; q < KNN; ++q) {
        if (cd > bd[q]) {
          float td = bd[q]; int ti = bi[q];
          bd[q] = cd; bi[q] = ci; cd = td; ci = ti;
        }
      }
    }
  }
  int* op = idx_out + ((size_t)b * NPTS + i) * KNN;
#pragma unroll
  for (int q = 0; q < KNN; ++q) op[q] = bi[q];
}

// ---------------------------------------------------------------------------
// Kernel 2: fused edge-features + conv1 + BN1 + lrelu + conv2 (WMMA f16) +
// BN2 + lrelu + max-over-k.  One block = (batch, 8 points) -> 160 rows = 10
// exact 16-row M-tiles.  conv2 is 10x8 tiles of v_wmma_f32_16x16x32_f16,
// K = 64 (2 k-steps).  All intermediates stay in LDS.
// ---------------------------------------------------------------------------
__global__ __launch_bounds__(256) void edgeconv_kernel(
    const float* __restrict__ x, const int* __restrict__ knn,
    const float* __restrict__ w1,
    const float* __restrict__ g1, const float* __restrict__ b1,
    const float* __restrict__ m1, const float* __restrict__ v1,
    const float* __restrict__ w2,
    const float* __restrict__ g2, const float* __restrict__ b2,
    const float* __restrict__ m2, const float* __restrict__ v2,
    _Float16* __restrict__ act2) {
  extern __shared__ char smem[];
  float*    s1t1  = (float*)smem;                 // 128 f32: scale|shift for bn1
  float*    s2t2  = s1t1 + 128;                   // 256 f32: scale|shift for bn2
  float*    edgef = s2t2 + 256;                   // 160*6 f32 edge features
  _Float16* act1h = (_Float16*)(edgef + 960);     // 160*64 f16 (A matrix)
  _Float16* w2h   = act1h + 160 * 64;             // 128*64 f16 (B matrix, [out][in])
  _Float16* c2buf = w2h + 128 * 64;               // 160*128 f16 post-act conv2

  const int tid = threadIdx.x;
  const int b = blockIdx.x / (NPTS / PTS);
  const int n0 = (blockIdx.x % (NPTS / PTS)) * PTS;
  const float* xb = x + (size_t)b * 3 * NPTS;

  if (tid < 64) {
    float s = g1[tid] * rsqrtf(v1[tid] + EPS);
    s1t1[tid] = s; s1t1[64 + tid] = b1[tid] - m1[tid] * s;
  } else if (tid < 192) {
    int c = tid - 64;
    float s = g2[c] * rsqrtf(v2[c] + EPS);
    s2t2[c] = s; s2t2[128 + c] = b2[c] - m2[c] * s;
  }
  for (int t = tid; t < 128 * 64; t += 256) w2h[t] = (_Float16)w2[t];
  if (tid < 160) {
    int p = tid / KNN, k = tid % KNN;
    int n = n0 + p;
    int j = knn[((size_t)b * NPTS + n) * KNN + k];
    float c0 = xb[n], c1 = xb[NPTS + n], c2 = xb[2 * NPTS + n];
    float* er = edgef + tid * 6;
    er[0] = xb[j] - c0; er[1] = xb[NPTS + j] - c1; er[2] = xb[2 * NPTS + j] - c2;
    er[3] = c0; er[4] = c1; er[5] = c2;
  }
  __syncthreads();

  // conv1: 160 rows x 64 channels, 6-wide dot; scalar VALU (tiny K), f16 out.
  for (int it = 0; it < 40; ++it) {
    int r = (tid >> 6) + it * 4;
    int o = tid & 63;
    const float* er = edgef + r * 6;
    const float* wr = w1 + o * 6;
    float acc = er[0] * wr[0] + er[1] * wr[1] + er[2] * wr[2] +
                er[3] * wr[3] + er[4] * wr[4] + er[5] * wr[5];
    acc = acc * s1t1[o] + s1t1[64 + o];
    acc = acc >= 0.f ? acc : 0.2f * acc;
    act1h[r * 64 + o] = (_Float16)acc;
  }
  __syncthreads();

  // conv2 via WMMA: wave w owns N-tile w (16 out channels), loops 10 M-tiles.
  const int wave = tid >> 5;
  const int lane = tid & 31;
  const int col = lane & 15;
  const int hiHalf = lane >> 4;          // lane 16-31 half (K-offset 16, contiguous k order for A and B)
  const int ch = wave * 16 + col;        // out channel of this lane's C columns
  const float s2 = s2t2[ch], t2 = s2t2[128 + ch];

  v16h bf[2];
#pragma unroll
  for (int ks = 0; ks < 2; ++ks)
    bf[ks] = *(const v16h*)(w2h + ch * 64 + ks * 32 + hiHalf * 16);

  for (int mt = 0; mt < 10; ++mt) {
    const _Float16* ap = act1h + (mt * 16 + col) * 64;
    v8f c = {};
#pragma unroll
    for (int ks = 0; ks < 2; ++ks) {
      int off = ks * 32 + hiHalf * 16;
      v8h lo = *(const v8h*)(ap + off);
      v8h hi = *(const v8h*)(ap + off + 8);
      v16h a;
#pragma unroll
      for (int q = 0; q < 8; ++q) { a[q] = lo[q]; a[q + 8] = hi[q]; }
      c = __builtin_amdgcn_wmma_f32_16x16x32_f16(false, a, false, bf[ks],
                                                 (short)0, c, false, false);
    }
#pragma unroll
    for (int v = 0; v < 8; ++v) {
      int r = mt * 16 + v + hiHalf * 8;   // C layout: M = v (+8 for hi lanes)
      float val = c[v] * s2 + t2;
      val = val >= 0.f ? val : 0.2f * val;
      c2buf[r * 128 + ch] = (_Float16)val;
    }
  }
  __syncthreads();

  // max over k per (point, channel), write f16 activation for conv3
  for (int t = tid; t < PTS * 128; t += 256) {
    int p = t >> 7, cch = t & 127;
    float mx = (float)c2buf[(p * KNN) * 128 + cch];
#pragma unroll
    for (int k = 1; k < KNN; ++k)
      mx = fmaxf(mx, (float)c2buf[(p * KNN + k) * 128 + cch]);
    act2[((size_t)b * NPTS + n0 + p) * 128 + cch] = (_Float16)mx;
  }
}

// ---------------------------------------------------------------------------
// Kernel 3: fused conv3 (128->1024, WMMA f16) + BN3 + lrelu + max over all
// 2048 points.  One block = (batch, 128-channel block); wave w owns 16
// channels, preloads its 4 B-fragments once, streams 128 M-tiles of n,
// folding the row-max into registers (shfl_xor merges the two C halves).
// Only [B,1024] f32 ever reaches HBM.
// ---------------------------------------------------------------------------
__global__ __launch_bounds__(256) void conv3_kernel(
    const _Float16* __restrict__ act2, const float* __restrict__ w3,
    const float* __restrict__ g3, const float* __restrict__ b3,
    const float* __restrict__ m3, const float* __restrict__ v3,
    float* __restrict__ feat) {
  __shared__ __align__(32) _Float16 w3h[128 * 128];
  const int tid = threadIdx.x;
  const int b = blockIdx.x >> 3;
  const int cb = (blockIdx.x & 7) * 128;
  for (int t = tid; t < 128 * 128; t += 256)
    w3h[t] = (_Float16)w3[(size_t)(cb + (t >> 7)) * 128 + (t & 127)];
  __syncthreads();

  const int wave = tid >> 5, lane = tid & 31;
  const int col = lane & 15, hiHalf = lane >> 4;
  const int lch = wave * 16 + col;
  const int ch = cb + lch;
  const float s3 = g3[ch] * rsqrtf(v3[ch] + EPS);
  const float t3 = b3[ch] - m3[ch] * s3;

  v16h bf[4];
#pragma unroll
  for (int ks = 0; ks < 4; ++ks)
    bf[ks] = *(const v16h*)(w3h + lch * 128 + ks * 32 + hiHalf * 16);

  const _Float16* ab = act2 + (size_t)b * NPTS * 128;
  float run = -3.4e38f;
  for (int mt = 0; mt < NPTS / 16; ++mt) {
    const _Float16* ap = ab + (size_t)(mt * 16 + col) * 128;
    v8f c = {};
#pragma unroll
    for (int ks = 0; ks < 4; ++ks) {
      int off = ks * 32 + hiHalf * 16;
      v8h lo = *(const v8h*)(ap + off);
      v8h hi = *(const v8h*)(ap + off + 8);
      v16h a;
#pragma unroll
      for (int q = 0; q < 8; ++q) { a[q] = lo[q]; a[q + 8] = hi[q]; }
      c = __builtin_amdgcn_wmma_f32_16x16x32_f16(false, a, false, bf[ks],
                                                 (short)0, c, false, false);
    }
    float tm = -3.4e38f;
#pragma unroll
    for (int v = 0; v < 8; ++v) {
      float val = c[v] * s3 + t3;
      val = val >= 0.f ? val : 0.2f * val;
      tm = fmaxf(tm, val);
    }
    run = fmaxf(run, tm);
  }
  run = fmaxf(run, __shfl_xor(run, 16, 32));   // merge M=v with M=v+8 half
  if (hiHalf == 0) feat[(size_t)b * 1024 + ch] = run;
}

// ---------------------------------------------------------------------------
// FC head (tiny): one thread per (b, out) dot product + BN + lrelu.
// ---------------------------------------------------------------------------
__global__ void fc_bn_kernel(const float* __restrict__ in, const float* __restrict__ w,
                             const float* __restrict__ g, const float* __restrict__ bb,
                             const float* __restrict__ m, const float* __restrict__ v,
                             float* __restrict__ out, int IN, int OUT) {
  int t = blockIdx.x * blockDim.x + threadIdx.x;
  if (t >= NB * OUT) return;
  int b = t / OUT, o = t % OUT;
  const float* ir = in + (size_t)b * IN;
  const float* wr = w + (size_t)o * IN;
  float acc = 0.f;
  for (int k = 0; k < IN; ++k) acc += ir[k] * wr[k];
  float s = g[o] * rsqrtf(v[o] + EPS);
  acc = acc * s + (bb[o] - m[o] * s);
  out[t] = acc >= 0.f ? acc : 0.2f * acc;
}

__global__ void tmat_kernel(const float* __restrict__ h2, const float* __restrict__ wt,
                            const float* __restrict__ bt, float* __restrict__ T) {
  int t = threadIdx.x;
  if (t >= NB * 9) return;
  int b = t / 9, i = t % 9;
  const float* hr = h2 + b * 256;
  const float* wr = wt + i * 256;
  float acc = bt[i];
  for (int k = 0; k < 256; ++k) acc += hr[k] * wr[k];
  T[t] = acc;
}

__global__ void apply_kernel(const float* __restrict__ x, const float* __restrict__ T,
                             float* __restrict__ out) {
  int t = blockIdx.x * blockDim.x + threadIdx.x;  // b*2048 + n
  int b = t >> 11, n = t & (NPTS - 1);
  const float* xb = x + (size_t)b * 3 * NPTS;
  float x0 = xb[n], x1 = xb[NPTS + n], x2 = xb[2 * NPTS + n];
  const float* Tb = T + b * 9;
  float* ob = out + (size_t)b * 3 * NPTS;
#pragma unroll
  for (int d = 0; d < 3; ++d)
    ob[d * NPTS + n] = x0 * Tb[d] + x1 * Tb[3 + d] + x2 * Tb[6 + d];
}

extern "C" void kernel_launch(void* const* d_in, const int* in_sizes, int n_in,
                              void* d_out, int out_size, void* d_ws, size_t ws_size,
                              hipStream_t stream) {
  const float* x   = (const float*)d_in[0];
  const float* w1  = (const float*)d_in[1];
  const float* g1  = (const float*)d_in[2];
  const float* b1  = (const float*)d_in[3];
  const float* m1  = (const float*)d_in[4];
  const float* v1  = (const float*)d_in[5];
  const float* w2  = (const float*)d_in[6];
  const float* g2  = (const float*)d_in[7];
  const float* b2  = (const float*)d_in[8];
  const float* m2  = (const float*)d_in[9];
  const float* v2  = (const float*)d_in[10];
  const float* w3  = (const float*)d_in[11];
  const float* g3  = (const float*)d_in[12];
  const float* b3  = (const float*)d_in[13];
  const float* m3  = (const float*)d_in[14];
  const float* v3  = (const float*)d_in[15];
  const float* wl1 = (const float*)d_in[16];
  const float* g4  = (const float*)d_in[17];
  const float* b4  = (const float*)d_in[18];
  const float* m4  = (const float*)d_in[19];
  const float* v4  = (const float*)d_in[20];
  const float* wl2 = (const float*)d_in[21];
  const float* g5  = (const float*)d_in[22];
  const float* b5  = (const float*)d_in[23];
  const float* m5  = (const float*)d_in[24];
  const float* v5  = (const float*)d_in[25];
  const float* wt  = (const float*)d_in[26];
  const float* bt  = (const float*)d_in[27];

  char* ws = (char*)d_ws;
  size_t off = 0;
  int* knn = (int*)(ws + off);          off += (size_t)NB * NPTS * KNN * 4;   // 5.24 MB
  _Float16* act2 = (_Float16*)(ws + off); off += (size_t)NB * NPTS * 128 * 2; // 16.78 MB
  float* feat = (float*)(ws + off);     off += (size_t)NB * 1024 * 4;
  float* h1 = (float*)(ws + off);       off += (size_t)NB * 512 * 4;
  float* h2 = (float*)(ws + off);       off += (size_t)NB * 256 * 4;
  float* T  = (float*)(ws + off);       off += (size_t)NB * 9 * 4;

  knn_kernel<<<NB * (NPTS / 256), 256, 0, stream>>>(x, knn);

  size_t smem2 = (128 + 256 + 960) * sizeof(float)      // bn params + edge feats
               + (160 * 64 + 128 * 64 + 160 * 128) * 2; // act1h + w2h + c2buf (f16)
  edgeconv_kernel<<<NB * (NPTS / PTS), 256, smem2, stream>>>(
      x, knn, w1, g1, b1, m1, v1, w2, g2, b2, m2, v2, act2);

  conv3_kernel<<<NB * 8, 256, 0, stream>>>(act2, w3, g3, b3, m3, v3, feat);

  fc_bn_kernel<<<(NB * 512 + 255) / 256, 256, 0, stream>>>(feat, wl1, g4, b4, m4, v4,
                                                           h1, 1024, 512);
  fc_bn_kernel<<<(NB * 256 + 255) / 256, 256, 0, stream>>>(h1, wl2, g5, b5, m5, v5,
                                                           h2, 512, 256);
  tmat_kernel<<<1, 288, 0, stream>>>(h2, wt, bt, T);
  apply_kernel<<<NB * NPTS / 256, 256, 0, stream>>>(x, T, (float*)d_out);
}
